// ColBlockQuantizedLinear_50457275793412
// MI455X (gfx1250) — matrix operand order repaired and checked by reading, not compiled, not run on hardware
//
#include <hip/hip_runtime.h>

// ColBlockQuantizedLinear for MI455X (gfx1250):
//   out[M,N] = inp[M,K] x dequant(qweight[N,K/2], scales[N], zeros[N])^T
//   M=8192, K=4096, N=11008.  Memory-bound (~590MB @ 23.3TB/s).
// Strategy: 128x128 C tile / 256 threads (8 waves), BK=32, fused nibble
// dequant into LDS f16 staging, v_wmma_f32_16x16x32_f16 math, NT weight
// loads + NT output stores, A kept L2-resident via N-fast rasterization.

typedef __attribute__((ext_vector_type(16))) _Float16 v16h;
typedef __attribute__((ext_vector_type(8)))  _Float16 v8h;
typedef __attribute__((ext_vector_type(4)))  _Float16 v4h;
typedef __attribute__((ext_vector_type(2)))  _Float16 v2h;
typedef __attribute__((ext_vector_type(8)))  float    v8f;
typedef __attribute__((ext_vector_type(4)))  float    vf4;
typedef __attribute__((ext_vector_type(4)))  int      vi4;

#define K_DIM   4096
#define N_DIM   11008
#define BM      128
#define BN      128
#define BK      32
#define LDS_STR 40              // 32 + 8 halves pad -> conflict-free b128 frag loads
#define NTHREADS 256

__global__ __launch_bounds__(NTHREADS)
void qlinear4_wmma_kernel(const float* __restrict__ A,   // [M, K] fp32
                          const int*   __restrict__ Q,   // [N, K/2] packed nibbles in int32
                          const float* __restrict__ S,   // [N] scales
                          const float* __restrict__ Z,   // [N] zeros
                          float*       __restrict__ C,   // [M, N] fp32
                          int M) {
    __shared__ _Float16 lA[BM][LDS_STR];
    __shared__ _Float16 lB[BN][LDS_STR];

    const int tid  = threadIdx.x;
    const int lane = tid & 31;
    const int wave = tid >> 5;
    const int waveM = wave & 3;   // 4 waves along M: subtiles {2*waveM, 2*waveM+1}
    const int waveN = wave >> 2;  // 2 waves along N: subtiles {4*waveN .. +3}

    const int bm = blockIdx.y * BM;      // M panel
    const int bn = blockIdx.x * BN;      // N panel (fast grid dim -> A stays in L2)

    // ---- global->LDS staging assignments -------------------------------
    // A tile (128 rows x 32 f32): 4 passes of 32 rows; thread t: row=t/8, 4 floats.
    const int arow = tid >> 3;           // 0..31
    const int acol = (tid & 7) << 2;     // 0,4,...,28
    // B tile (128 rows x 16 words): thread t: row=t/2, 8 words (two int4).
    const int brow = tid >> 1;           // 0..127
    const int bw0  = (tid & 1) << 3;     // word offset 0 or 8

    const float scale = S[bn + brow];
    const float zero  = Z[bn + brow];

    const int KW = K_DIM >> 1;           // 2048 words per weight row

    v8f acc[2][4] = {};

    vf4 aReg[4];
    vi4 qReg[2];

    auto loadGlobal = [&](int k0) {
        #pragma unroll
        for (int p = 0; p < 4; ++p) {
            const float* pa = A + (size_t)(bm + p * 32 + arow) * K_DIM + k0 + acol;
            aReg[p] = *(const vf4*)pa;                       // RT: want L2 retention
        }
        const int* pq = Q + (size_t)(bn + brow) * KW + (k0 >> 1) + bw0;
        qReg[0] = __builtin_nontemporal_load((const vi4*)pq);       // streamed once
        qReg[1] = __builtin_nontemporal_load((const vi4*)pq + 1);
    };

    auto storeLds = [&]() {
        #pragma unroll
        for (int p = 0; p < 4; ++p) {
            v4h t;
            t[0] = (_Float16)aReg[p][0];
            t[1] = (_Float16)aReg[p][1];
            t[2] = (_Float16)aReg[p][2];
            t[3] = (_Float16)aReg[p][3];
            *(v4h*)&lA[p * 32 + arow][acol] = t;
        }
        #pragma unroll
        for (int u = 0; u < 2; ++u) {
            #pragma unroll
            for (int j = 0; j < 4; ++j) {
                const int q  = qReg[u][j];
                const int wi = bw0 + u * 4 + j;             // word idx in tile: 0..15
                v2h t;
                t[0] = (_Float16)(((float)(q & 15))        - zero) * (_Float16)0 + // keep fp32 math:
                       (_Float16)((((float)(q & 15))       - zero) * scale);
                t[1] = (_Float16)((((float)((q >> 4) & 15)) - zero) * scale);
                // note: t[0] expression above simplifies to the dequant value; the
                // zero-multiplied term is elided by the compiler (kept for clarity
                // that conversion happens after fp32 dequant arithmetic).
                t[0] = (_Float16)((((float)(q & 15)) - zero) * scale);
                *(v2h*)&lB[brow][wi * 2] = t;
            }
        }
    };

    auto compute = [&]() {
        union HV { v8h h8[2]; v16h h16; };
        v16h af[2], bf[4];
        const int m   = lane & 15;
        const int kbA = (lane >> 4) << 3;    // ISA 16-bit A 16x32 layout: 0 or 8
        #pragma unroll
        for (int i = 0; i < 2; ++i) {
            const _Float16* pa = &lA[(waveM * 2 + i) * 16 + m][kbA];
            HV t;
            t.h8[0] = *(const v8h*)pa;          // K = kbA..kbA+7
            t.h8[1] = *(const v8h*)(pa + 16);   // K = kbA+16..kbA+23
            af[i] = t.h16;
        }
        const int kbB = (lane >> 4) << 3;    // ISA 16-bit B 32x16 layout: 0 or 8
        #pragma unroll
        for (int j = 0; j < 4; ++j) {
            const _Float16* pb = &lB[(waveN * 4 + j) * 16 + m][kbB];
            HV t;
            t.h8[0] = *(const v8h*)pb;          // K = kbB..kbB+7
            t.h8[1] = *(const v8h*)(pb + 16);   // K = kbB+16..kbB+23
            bf[j] = t.h16;
        }
        #pragma unroll
        for (int i = 0; i < 2; ++i)
            #pragma unroll
            for (int j = 0; j < 4; ++j)
                acc[i][j] = __builtin_amdgcn_wmma_f32_16x16x32_f16(
                    /*neg_a=*/false, af[i], /*neg_b=*/false, bf[j],
                    /*c_mod=*/(short)0, acc[i][j],
                    /*reuse_a=*/false, /*reuse_b=*/false);
    };

    // ---- main K loop: store->barrier->prefetch-next->compute ------------
    loadGlobal(0);
    for (int k0 = 0; k0 < K_DIM; k0 += BK) {
        __syncthreads();
        storeLds();
        __syncthreads();
        if (k0 + BK < K_DIM) loadGlobal(k0 + BK);   // overlap DRAM with WMMA
        compute();
    }

    // ---- writeback (NT: output never re-read) ---------------------------
    const int n  = lane & 15;
    const int mb = (lane >> 4) << 3;    // C/D layout: lanes 16-31 hold M=8..15
    #pragma unroll
    for (int i = 0; i < 2; ++i) {
        #pragma unroll
        for (int j = 0; j < 4; ++j) {
            const int row0 = bm + (waveM * 2 + i) * 16 + mb;
            const int col  = bn + (waveN * 4 + j) * 16 + n;
            #pragma unroll
            for (int v = 0; v < 8; ++v) {
                __builtin_nontemporal_store(acc[i][j][v],
                    C + (size_t)(row0 + v) * N_DIM + col);
            }
        }
    }
}

extern "C" void kernel_launch(void* const* d_in, const int* in_sizes, int n_in,
                              void* d_out, int out_size, void* d_ws, size_t ws_size,
                              hipStream_t stream) {
    const float* inp    = (const float*)d_in[0];   // [4,2048,4096] fp32
    const int*   qw     = (const int*)  d_in[1];   // [11008,2048] int32 (byte values)
    const float* scales = (const float*)d_in[2];   // [11008]
    const float* zeros  = (const float*)d_in[3];   // [11008]
    float*       out    = (float*)d_out;           // [4,2048,11008] fp32

    const int M = in_sizes[0] / K_DIM;             // 8192

    dim3 grid(N_DIM / BN, M / BM);                 // (86, 64); N fast -> A L2-resident
    dim3 block(NTHREADS);
    qlinear4_wmma_kernel<<<grid, block, 0, stream>>>(inp, qw, scales, zeros, out, M);
}
